// RGATSQL_21457656611019
// MI455X (gfx1250) — matrix operand order repaired and checked by reading, not compiled, not run on hardware
//
#include <hip/hip_runtime.h>
#include <hip/hip_bf16.h>

typedef __attribute__((ext_vector_type(16))) _Float16 v16h;
typedef __attribute__((ext_vector_type(8)))  _Float16 v8h;
typedef __attribute__((ext_vector_type(8)))  float    v8f;

#define N_NODES 20000
#define N_EDGES 640000
#define HID     256
#define HEADS   8
#define DK      32
#define FFN_HID 1024

// ---------------------------------------------------------------------------
// f32 -> f16 conversion (row-major, for activations)
// ---------------------------------------------------------------------------
__global__ __launch_bounds__(256)
void cvt_f16_kernel(const float* __restrict__ in, _Float16* __restrict__ out, int n) {
    int i = blockIdx.x * 256 + threadIdx.x;
    if (i < n) out[i] = (_Float16)in[i];
}

// f32 [K,N] -> f16 [N,K] (transpose) for weights: makes WMMA B-fragments
// 16 contiguous halves per lane.
__global__ __launch_bounds__(256)
void cvt_tr_kernel(const float* __restrict__ in, _Float16* __restrict__ out, int K, int N) {
    int i = blockIdx.x * 256 + threadIdx.x;
    if (i >= K * N) return;
    int k = i / N;
    int n = i - k * N;
    out[(size_t)n * K + k] = (_Float16)in[i];
}

__global__ __launch_bounds__(256)
void zero_f32_kernel(float* __restrict__ p, int n) {
    int i = blockIdx.x * 256 + threadIdx.x;
    if (i < n) p[i] = 0.0f;
}

// ---------------------------------------------------------------------------
// WMMA GEMM: C[M,N] = A[M,K](f16,row-major) @ Bt[N,K](f16,row-major=B^T)
// One wave computes a 16(M) x 64(N) tile: 4 f32 accumulators, A fragment
// reused across 4 v_wmma_f32_16x16x32_f16 per K-step. No LDS, no barriers.
// mode: 0 = +bias (f32 out), 1 = +bias, relu (f16 out), 2 = no bias (f32 out)
// Requires M%16==0, N%64==0, K%32==0.
// ---------------------------------------------------------------------------
__global__ __launch_bounds__(256)
void gemm_wmma_kernel(const _Float16* __restrict__ A,
                      const _Float16* __restrict__ Bt,
                      const float*    __restrict__ bias,
                      float*          __restrict__ Cf,
                      _Float16*       __restrict__ Ch,
                      int M, int N, int K, int mode) {
    const int lane   = threadIdx.x & 31;
    const int waveId = (int)blockIdx.x * (256 >> 5) + ((int)threadIdx.x >> 5);
    const int tilesN = N >> 6;
    const int tilesM = M >> 4;
    if (waveId >= tilesM * tilesN) return;
    const int tileM = (waveId / tilesN) * 16;
    const int tileN = (waveId % tilesN) * 64;

    const int m16   = lane & 15;     // row within A tile / col within subtile
    const int half  = lane >> 4;     // K half selector (A and B: k 0..15 vs 16..31)

    // A: lane reads row (tileM+m16); 16 contiguous halves at k0+half*16
    const _Float16* aRow = A + (size_t)(tileM + m16) * K;
    // B fragment: lane holds column n = s*16+m16, K-range khalf*16..+15 -> contiguous in Bt
    const _Float16* bRow0 = Bt + (size_t)(tileN + 0 * 16 + m16) * K + half * 16;
    const _Float16* bRow1 = Bt + (size_t)(tileN + 1 * 16 + m16) * K + half * 16;
    const _Float16* bRow2 = Bt + (size_t)(tileN + 2 * 16 + m16) * K + half * 16;
    const _Float16* bRow3 = Bt + (size_t)(tileN + 3 * 16 + m16) * K + half * 16;

    v8f acc0 = {}, acc1 = {}, acc2 = {}, acc3 = {};

    for (int k0 = 0; k0 < K; k0 += 32) {
        __builtin_prefetch(aRow + k0 + 64, 0, 1);

        v16h a = *(const v16h*)(aRow + k0 + half * 16);

        v16h b0 = *(const v16h*)(bRow0 + k0);
        v16h b1 = *(const v16h*)(bRow1 + k0);
        v16h b2 = *(const v16h*)(bRow2 + k0);
        v16h b3 = *(const v16h*)(bRow3 + k0);

        acc0 = __builtin_amdgcn_wmma_f32_16x16x32_f16(false, a, false, b0, (short)0, acc0, false, false);
        acc1 = __builtin_amdgcn_wmma_f32_16x16x32_f16(false, a, false, b1, (short)0, acc1, false, false);
        acc2 = __builtin_amdgcn_wmma_f32_16x16x32_f16(false, a, false, b2, (short)0, acc2, false, false);
        acc3 = __builtin_amdgcn_wmma_f32_16x16x32_f16(false, a, false, b3, (short)0, acc3, false, false);
    }

    // Epilogue. C/D layout: VGPR j -> row (lane<16 ? j : 8+j), col = lane&15.
    const int rbase = tileM + half * 8;
    v8f accs[4] = {acc0, acc1, acc2, acc3};
#pragma unroll
    for (int s = 0; s < 4; ++s) {
        const int col = tileN + s * 16 + m16;
        const float bv = (mode != 2) ? bias[col] : 0.0f;
#pragma unroll
        for (int j = 0; j < 8; ++j) {
            float v = accs[s][j] + bv;
            if (mode == 1) {
                v = fmaxf(v, 0.0f);
                Ch[(size_t)(rbase + j) * N + col] = (_Float16)v;
            } else {
                Cf[(size_t)(rbase + j) * N + col] = v;
            }
        }
    }
}

// ---------------------------------------------------------------------------
// Edge attention: one block (8 waves) per edge batch; wave == head (DK==32).
// score = sum((k[src]+e) * q[dst]); s = exp(clamp(score/sqrt(32),-10,10))
// wv[dst] += (v[src]+e)*s ; z[dst,h] += s
// ---------------------------------------------------------------------------
#define EDGES_PER_BLOCK 8

__global__ __launch_bounds__(256)
void edge_attn_kernel(const float* __restrict__ q, const float* __restrict__ k,
                      const float* __restrict__ v, const float* __restrict__ rel,
                      const int* __restrict__ edges, const int* __restrict__ src,
                      const int* __restrict__ dst,
                      float* __restrict__ wv, float* __restrict__ z, int E) {
    const int t    = threadIdx.x;
    const int head = t >> 5;
    const int d    = t & 31;
    const float inv_scale = 0.17677669529663689f;  // 1/sqrt(32)

    int e0 = blockIdx.x * EDGES_PER_BLOCK;
    for (int i = 0; i < EDGES_PER_BLOCK; ++i) {
        int e = e0 + i;
        if (e >= E) break;
        const int sn = src[e];
        const int dn = dst[e];
        const int r  = edges[e];

        const float eh = rel[r * DK + d];
        const float kk = k[(size_t)sn * HID + t];
        const float qq = q[(size_t)dn * HID + t];

        float prod = (kk + eh) * qq;
#pragma unroll
        for (int off = 16; off > 0; off >>= 1)
            prod += __shfl_xor(prod, off, 32);

        float sc = prod * inv_scale;
        sc = fminf(fmaxf(sc, -10.0f), 10.0f);
        const float s = __expf(sc);

        const float vv = v[(size_t)sn * HID + t];
        atomicAdd(&wv[(size_t)dn * HID + t], (vv + eh) * s);
        if (d == 0) atomicAdd(&z[dn * HEADS + head], s);
    }
}

// ---------------------------------------------------------------------------
// o = wv / max(z,1e-8), converted to f16 for the O-projection GEMM
// ---------------------------------------------------------------------------
__global__ __launch_bounds__(256)
void onorm_cvt_kernel(const float* __restrict__ wv, const float* __restrict__ z,
                      _Float16* __restrict__ of16, int n) {
    int i = blockIdx.x * 256 + threadIdx.x;
    if (i >= n) return;
    int node = i >> 8;
    int head = (i >> 5) & 7;
    float zz = fmaxf(z[node * HEADS + head], 1e-8f);
    of16[i] = (_Float16)(wv[i] / zz);
}

// ---------------------------------------------------------------------------
// out = LayerNorm(a + b) * g + beta ; one wave per 256-wide row
// ---------------------------------------------------------------------------
__global__ __launch_bounds__(256)
void add_ln_kernel(const float* __restrict__ a, const float* __restrict__ b,
                   const float* __restrict__ g, const float* __restrict__ beta,
                   float* __restrict__ out, int rows) {
    const int wave = threadIdx.x >> 5;
    const int lane = threadIdx.x & 31;
    const int row  = blockIdx.x * 8 + wave;
    if (row >= rows) return;

    const float* ar = a + (size_t)row * HID;
    const float* br = b + (size_t)row * HID;

    float vals[8];
    float sum = 0.0f;
#pragma unroll
    for (int j = 0; j < 8; ++j) {
        float x = ar[lane + j * 32] + br[lane + j * 32];
        vals[j] = x;
        sum += x;
    }
#pragma unroll
    for (int off = 16; off > 0; off >>= 1) sum += __shfl_xor(sum, off, 32);
    const float mu = sum * (1.0f / HID);

    float var = 0.0f;
#pragma unroll
    for (int j = 0; j < 8; ++j) { float dx = vals[j] - mu; var += dx * dx; }
#pragma unroll
    for (int off = 16; off > 0; off >>= 1) var += __shfl_xor(var, off, 32);
    var *= (1.0f / HID);
    const float rstd = rsqrtf(var + 1e-5f);

#pragma unroll
    for (int j = 0; j < 8; ++j) {
        int c = lane + j * 32;
        out[(size_t)row * HID + c] = (vals[j] - mu) * rstd * g[c] + beta[c];
    }
}

// ---------------------------------------------------------------------------
// Host-side orchestration
// ---------------------------------------------------------------------------
static inline size_t align256(size_t x) { return (x + 255) & ~(size_t)255; }

extern "C" void kernel_launch(void* const* d_in, const int* in_sizes, int n_in,
                              void* d_out, int out_size, void* d_ws, size_t ws_size,
                              hipStream_t stream) {
    const float* x    = (const float*)d_in[0];
    const int*   edges= (const int*)  d_in[1];
    const int*   src  = (const int*)  d_in[2];
    const int*   dst  = (const int*)  d_in[3];
    const float* rel  = (const float*)d_in[4];
    const float* Wq   = (const float*)d_in[5];
    const float* bq   = (const float*)d_in[6];
    const float* Wk   = (const float*)d_in[7];
    const float* Wv   = (const float*)d_in[8];
    const float* Wo   = (const float*)d_in[9];
    const float* bo   = (const float*)d_in[10];
    const float* ln1g = (const float*)d_in[11];
    const float* ln1b = (const float*)d_in[12];
    const float* W1   = (const float*)d_in[13];
    const float* b1   = (const float*)d_in[14];
    const float* W2   = (const float*)d_in[15];
    const float* b2   = (const float*)d_in[16];
    const float* ln2g = (const float*)d_in[17];
    const float* ln2b = (const float*)d_in[18];

    const size_t NH  = (size_t)N_NODES * HID;       // 5.12M
    const size_t NF  = (size_t)N_NODES * FFN_HID;   // 20.48M
    const size_t WSQ = (size_t)HID * HID;           // 65536 per layer
    const size_t WS1 = (size_t)HID * FFN_HID;       // 262144 per layer

    char* base = (char*)d_ws;
    size_t off = 0;
    auto alloc = [&](size_t bytes) { size_t o = off; off += align256(bytes); return (void*)(base + o); };

    _Float16* wq16  = (_Float16*)alloc(2 * WSQ * sizeof(_Float16));  // [N,K] transposed
    _Float16* wk16  = (_Float16*)alloc(2 * WSQ * sizeof(_Float16));
    _Float16* wv16  = (_Float16*)alloc(2 * WSQ * sizeof(_Float16));
    _Float16* wo16  = (_Float16*)alloc(2 * WSQ * sizeof(_Float16));
    _Float16* w116  = (_Float16*)alloc(2 * WS1 * sizeof(_Float16));  // [1024,256]^T per layer
    _Float16* w216  = (_Float16*)alloc(2 * WS1 * sizeof(_Float16));  // [256,1024]^T per layer
    _Float16* xf16  = (_Float16*)alloc(NH * sizeof(_Float16));
    _Float16* hf16  = (_Float16*)alloc(NH * sizeof(_Float16));
    _Float16* f1f16 = (_Float16*)alloc(NF * sizeof(_Float16));
    float* qbuf  = (float*)alloc(NH * sizeof(float));
    float* kbuf  = (float*)alloc(NH * sizeof(float));
    float* vbuf  = (float*)alloc(NH * sizeof(float));
    float* wvbuf = (float*)alloc(NH * sizeof(float));
    float* zbuf  = (float*)alloc((size_t)N_NODES * HEADS * sizeof(float));
    float* tmp   = (float*)alloc(NH * sizeof(float));
    float* hbuf  = (float*)alloc(NH * sizeof(float));
    float* x1    = (float*)alloc(NH * sizeof(float));
    (void)ws_size;

    auto cvt = [&](const float* in, _Float16* out, size_t n) {
        cvt_f16_kernel<<<dim3((unsigned)((n + 255) / 256)), 256, 0, stream>>>(in, out, (int)n);
    };
    auto cvt_tr = [&](const float* in, _Float16* out, int K, int N) {
        int n = K * N;
        cvt_tr_kernel<<<dim3((unsigned)((n + 255) / 256)), 256, 0, stream>>>(in, out, K, N);
    };
    auto gemm = [&](const _Float16* A, const _Float16* Bt, const float* bias,
                    float* Cf, _Float16* Ch, int M, int N, int K, int mode) {
        int tiles = (M >> 4) * (N >> 6);
        gemm_wmma_kernel<<<dim3((unsigned)((tiles + 7) / 8)), 256, 0, stream>>>(
            A, Bt, bias, Cf, Ch, M, N, K, mode);
    };

    // one-time transposed weight conversion (both layers)
    for (int i = 0; i < 2; ++i) {
        cvt_tr(Wq + i * WSQ, wq16 + i * WSQ, HID, HID);
        cvt_tr(Wk + i * WSQ, wk16 + i * WSQ, HID, HID);
        cvt_tr(Wv + i * WSQ, wv16 + i * WSQ, HID, HID);
        cvt_tr(Wo + i * WSQ, wo16 + i * WSQ, HID, HID);
        cvt_tr(W1 + i * WS1, w116 + i * WS1, HID, FFN_HID);
        cvt_tr(W2 + i * WS1, w216 + i * WS1, FFN_HID, HID);
    }

    const float* xin = x;
    float* xouts[2] = { x1, (float*)d_out };

    for (int i = 0; i < 2; ++i) {
        const _Float16* wq_i = wq16 + (size_t)i * WSQ;
        const _Float16* wk_i = wk16 + (size_t)i * WSQ;
        const _Float16* wv_i = wv16 + (size_t)i * WSQ;
        const _Float16* wo_i = wo16 + (size_t)i * WSQ;
        const _Float16* w1_i = w116 + (size_t)i * WS1;
        const _Float16* w2_i = w216 + (size_t)i * WS1;
        const float* bq_i = bq + (size_t)i * HID;
        const float* bo_i = bo + (size_t)i * HID;
        const float* b1_i = b1 + (size_t)i * FFN_HID;
        const float* b2_i = b2 + (size_t)i * HID;

        // x -> f16
        cvt(xin, xf16, NH);

        // q/k/v projections (M=20000, N=256, K=256)
        gemm(xf16, wq_i, bq_i, qbuf, nullptr, N_NODES, HID, HID, 0);
        gemm(xf16, wk_i, nullptr, kbuf, nullptr, N_NODES, HID, HID, 2);
        gemm(xf16, wv_i, nullptr, vbuf, nullptr, N_NODES, HID, HID, 2);

        // zero segment-sum accumulators
        zero_f32_kernel<<<dim3((unsigned)((NH + 255) / 256)), 256, 0, stream>>>(wvbuf, (int)NH);
        zero_f32_kernel<<<dim3((N_NODES * HEADS + 255) / 256), 256, 0, stream>>>(zbuf, N_NODES * HEADS);

        // edge attention + segment sums
        edge_attn_kernel<<<dim3(N_EDGES / EDGES_PER_BLOCK), 256, 0, stream>>>(
            qbuf, kbuf, vbuf, rel, edges, src, dst, wvbuf, zbuf, N_EDGES);

        // o = wv/z -> f16 (reuse xf16 as o storage)
        onorm_cvt_kernel<<<dim3((unsigned)((NH + 255) / 256)), 256, 0, stream>>>(wvbuf, zbuf, xf16, (int)NH);

        // o @ Wo + bo -> tmp ; h = LN(x + tmp)
        gemm(xf16, wo_i, bo_i, tmp, nullptr, N_NODES, HID, HID, 0);
        add_ln_kernel<<<dim3(N_NODES / 8), 256, 0, stream>>>(xin, tmp,
            ln1g + (size_t)i * HID, ln1b + (size_t)i * HID, hbuf, N_NODES);

        // FFN: f1f16 = relu(h@W1+b1) in f16 directly; tmp = f1@W2+b2 ; x = LN(h+tmp)
        cvt(hbuf, hf16, NH);
        gemm(hf16, w1_i, b1_i, nullptr, f1f16, N_NODES, FFN_HID, HID, 1);
        gemm(f1f16, w2_i, b2_i, tmp, nullptr, N_NODES, HID, FFN_HID, 0);
        add_ln_kernel<<<dim3(N_NODES / 8), 256, 0, stream>>>(hbuf, tmp,
            ln2g + (size_t)i * HID, ln2b + (size_t)i * HID, xouts[i], N_NODES);

        xin = xouts[i];
    }
}
